// WindowAttention_14319420965432
// MI455X (gfx1250) — compile-verified, matrix operand order checked by reading, not run on hardware
//
#include <hip/hip_runtime.h>

// ---------------------------------------------------------------------------
// Swin window attention for MI455X (gfx1250, wave32, WMMA).
// One 128-thread block (4 waves) per window. All GEMMs via
// v_wmma_f32_16x16x32_bf16 (bf16 inputs, f32 accumulate), softmax in f32.
// Round 3: QKV column tiles remapped (t = ct*4 + wave) so the q/k/v scatter
// target is a compile-time constant -> three straight-line store paths, no
// EXEC-mask branches, no address selects. Rolled 2-iteration loops bound
// load staging so the allocation stays under 256 VGPRs (no s_set_vgpr_msb).
// ---------------------------------------------------------------------------

typedef __bf16 bf16_t;
typedef __attribute__((ext_vector_type(16))) __bf16 v16bf;
typedef __attribute__((ext_vector_type(8)))  __bf16 v8bf;
typedef __attribute__((ext_vector_type(8)))  float  v8f;

#define NTOK   64     // tokens per window (8x8)
#define DIMC   128    // channels
#define NH     4      // heads
#define HD     32     // head dim
#define SCALE  0.17677669529663687f  // 32^-0.5

// LDS layout (bytes). XB is reused as the attention-output buffer.
#define XB_STRIDE 136                 // halves per row (272 B, 16B-multiple)
#define QS_STRIDE 40                  // halves (80 B)
#define VT_STRIDE 72                  // halves (144 B)
#define PS_STRIDE 72
#define OFF_XB 0
#define OFF_QS 17408                  // 64*136*2
#define OFF_KS (OFF_QS + 4*64*QS_STRIDE*2)   // +20480
#define OFF_VT (OFF_KS + 4*64*QS_STRIDE*2)   // +20480
#define OFF_PS (OFF_VT + 4*32*VT_STRIDE*2)   // +18432
#define SMEM_BYTES (OFF_PS + 4*64*PS_STRIDE*2) // 113664

// Workspace layout (bytes)
#define WS_QKVW  0                     // 3*128*128 bf16 = 98304
#define WS_PROJW 98304                 // 128*128 bf16   = 32768
#define WS_BIAS  131072                // 4*64*64 f32    = 65536

// Fragment loader for 16-bit A/B operands of v_wmma_*_16x16x32.
// A: lane holds row m=(lane&15); halves 0..7 -> K = kg*8+0..7, halves 8..15
// -> K = 16+kg*8+0..7, kg = lane>>4.  B is symmetric (lane holds column n),
// so one helper serves both on [M][K]- / [N][K]-major storage.
__device__ __forceinline__ v16bf load_frag(const bf16_t* __restrict__ p,
                                           int stride, int row, int col, int lane) {
  int r  = row + (lane & 15);
  int kg = (lane >> 4) << 3;
  const v8bf* lo = (const v8bf*)(p + r * stride + col + kg);
  const v8bf* hi = (const v8bf*)(p + r * stride + col + 16 + kg);
  v8bf a = *lo, b = *hi;
  v16bf o;
#pragma unroll
  for (int h = 0; h < 8; ++h) { o[h] = a[h]; o[h + 8] = b[h]; }
  return o;
}

__device__ __forceinline__ v8f wmma_bf16(v16bf a, v16bf b, v8f c) {
  return __builtin_amdgcn_wmma_f32_16x16x32_bf16(false, a, false, b,
                                                 (short)0, c, false, false);
}

// ---------------------------------------------------------------------------
// Prep: bf16 weights + clamped relative-position bias (runs once per launch).
// ---------------------------------------------------------------------------
__global__ void wa_prep(const float* __restrict__ qkv_w,
                        const float* __restrict__ proj_w,
                        const float* __restrict__ rel_tbl,
                        bf16_t* __restrict__ qkv_wb,
                        bf16_t* __restrict__ proj_wb,
                        float*  __restrict__ bias_c) {
  int i = blockIdx.x * blockDim.x + threadIdx.x;
  if (i < 3 * DIMC * DIMC) qkv_wb[i] = (bf16_t)qkv_w[i];
  if (i < DIMC * DIMC)     proj_wb[i] = (bf16_t)proj_w[i];
  if (i < NH * NTOK * NTOK) {
    int h = i >> 12, ij = i & 4095, ii = ij >> 6, jj = ij & 63;
    int rel = ((ii >> 3) - (jj >> 3) + 7) * 15 + ((ii & 7) - (jj & 7) + 7);
    float bv = rel_tbl[rel * NH + h];
    bias_c[i] = fminf(5.f, fmaxf(-5.f, bv));
  }
}

// ---------------------------------------------------------------------------
// Main: one block per window.
// ---------------------------------------------------------------------------
__global__ __launch_bounds__(128)
void wa_main(const float* __restrict__ x,        // [4096][64][128]
             const float* __restrict__ mask,     // [64][64][64]
             const bf16_t* __restrict__ qkv_wb,  // [384][128] bf16
             const float* __restrict__ qkv_b,    // [384]
             const bf16_t* __restrict__ proj_wb, // [128][128] bf16
             const float* __restrict__ proj_b,   // [128]
             const float* __restrict__ bias_c,   // [4][64][64] clamped
             float* __restrict__ out)            // [4096][64][128]
{
  __shared__ __align__(16) char smem[SMEM_BYTES];
  bf16_t* XB = (bf16_t*)(smem + OFF_XB);   // x (bf16), later attn output
  bf16_t* QS = (bf16_t*)(smem + OFF_QS);   // [4][64][40]
  bf16_t* KS = (bf16_t*)(smem + OFF_KS);   // [4][64][40]
  bf16_t* VT = (bf16_t*)(smem + OFF_VT);   // [4][32][72]  (V transposed)
  bf16_t* PS = (bf16_t*)(smem + OFF_PS);   // [4][64][72]

  const int b    = blockIdx.x;
  const int tid  = threadIdx.x;
  const int wave = tid >> 5;
  const int lane = tid & 31;

  // Prefetch the x tile of a window ~96 blocks ahead into L2/WGP cache
  // (32 KB = 256 lines of 128 B; 2 lines per thread).
  {
    const int pb = b + 96;
    if (pb < (int)gridDim.x) {
      const char* p = (const char*)(x + (size_t)pb * NTOK * DIMC);
      __builtin_prefetch(p + tid * 128, 0, 0);
      __builtin_prefetch(p + 16384 + tid * 128, 0, 0);
    }
  }

  // ---- Stage A: x[b] -> bf16 LDS -----------------------------------------
  const float* xg = x + (size_t)b * NTOK * DIMC;
#pragma unroll 4
  for (int it = 0; it < 16; ++it) {
    int idx4 = it * 128 + tid;              // 0..2047 float4s
    float4 v = ((const float4*)xg)[idx4];
    int e = idx4 << 2;
    bf16_t* d = &XB[(e >> 7) * XB_STRIDE + (e & 127)];
    d[0] = (bf16_t)v.x; d[1] = (bf16_t)v.y; d[2] = (bf16_t)v.z; d[3] = (bf16_t)v.w;
  }
  __syncthreads();

  // ---- Stage B: QKV = x @ qkv_w^T + b; q *= SCALE; V stored transposed ----
  // Column tile t = ct*4 + wave, o0 = t*16  =>  which = ct>>1 is a
  // compile-time constant per loop: ct 0-1 -> Q, 2-3 -> K, 4-5 -> V.
#pragma unroll 1
  for (int mt = 0; mt < 4; ++mt) {
    const int m0   = mt * 16;
    const int mrow = m0 + ((lane >> 4) << 3);
    v16bf afr[4];
#pragma unroll
    for (int ks = 0; ks < 4; ++ks) afr[ks] = load_frag(XB, XB_STRIDE, m0, ks * 32, lane);

    // ---- Q tiles (which = 0): scale and store [head][m][d] --------------
#pragma unroll 1
    for (int ct = 0; ct < 2; ++ct) {
      const int t = ct * 4 + wave, o0 = t * 16;
      v8f acc = {};
#pragma unroll
      for (int ks = 0; ks < 4; ++ks)
        acc = wmma_bf16(afr[ks], load_frag(qkv_wb, DIMC, o0, ks * 32, lane), acc);
      const int   o    = o0 + (lane & 15);
      const int   head = (t >> 1) & 3;
      const int   d    = o & 31;
      const float bia  = qkv_b[o];
      bf16_t* dst = QS + head * 64 * QS_STRIDE + mrow * QS_STRIDE + d;
#pragma unroll
      for (int r = 0; r < 8; ++r) dst[r * QS_STRIDE] = (bf16_t)((acc[r] + bia) * SCALE);
    }
    // ---- K tiles (which = 1): store [head][m][d] ------------------------
#pragma unroll 1
    for (int ct = 2; ct < 4; ++ct) {
      const int t = ct * 4 + wave, o0 = t * 16;
      v8f acc = {};
#pragma unroll
      for (int ks = 0; ks < 4; ++ks)
        acc = wmma_bf16(afr[ks], load_frag(qkv_wb, DIMC, o0, ks * 32, lane), acc);
      const int   o    = o0 + (lane & 15);
      const int   head = (t >> 1) & 3;
      const int   d    = o & 31;
      const float bia  = qkv_b[o];
      bf16_t* dst = KS + head * 64 * QS_STRIDE + mrow * QS_STRIDE + d;
#pragma unroll
      for (int r = 0; r < 8; ++r) dst[r * QS_STRIDE] = (bf16_t)(acc[r] + bia);
    }
    // ---- V tiles (which = 2): store transposed [head][d][m] -------------
#pragma unroll 1
    for (int ct = 4; ct < 6; ++ct) {
      const int t = ct * 4 + wave, o0 = t * 16;
      v8f acc = {};
#pragma unroll
      for (int ks = 0; ks < 4; ++ks)
        acc = wmma_bf16(afr[ks], load_frag(qkv_wb, DIMC, o0, ks * 32, lane), acc);
      const int   o    = o0 + (lane & 15);
      const int   head = (t >> 1) & 3;
      const int   d    = o & 31;
      const float bia  = qkv_b[o];
      bf16_t* dst = VT + head * 32 * VT_STRIDE + d * VT_STRIDE + mrow;
#pragma unroll
      for (int r = 0; r < 8; ++r) dst[r] = (bf16_t)(acc[r] + bia);
    }
  }
  __syncthreads();

  // ---- Stage C: attention, one wave per head -----------------------------
  {
    const int h = wave;
    const bf16_t* Qh = QS + h * 64 * QS_STRIDE;
    const bf16_t* Kh = KS + h * 64 * QS_STRIDE;
    const bf16_t* Vh = VT + h * 32 * VT_STRIDE;
    bf16_t*       Ph = PS + h * 64 * PS_STRIDE;
    const float* maskw = mask + (size_t)(b & 63) * NTOK * NTOK;
    const float* biash = bias_c + h * NTOK * NTOK;

#pragma unroll 1
    for (int mt = 0; mt < 4; ++mt) {
      const int m0 = mt * 16;
      v16bf qf = load_frag(Qh, QS_STRIDE, m0, 0, lane);
      v8f S[4];
#pragma unroll
      for (int jt = 0; jt < 4; ++jt) {
        v16bf kf = load_frag(Kh, QS_STRIDE, jt * 16, 0, lane);
        v8f z = {};
        S[jt] = wmma_bf16(qf, kf, z);      // S = (q*SCALE) @ K^T
      }
      // + window mask + clamped rel-pos bias, clip to +-10
      const int i_lo = m0 + ((lane >> 4) << 3);
      const int jlan = lane & 15;
#pragma unroll
      for (int r = 0; r < 8; ++r) {
        const int i = i_lo + r;
#pragma unroll
        for (int jt = 0; jt < 4; ++jt) {
          const int e = i * 64 + jt * 16 + jlan;
          float val = S[jt][r] + maskw[e] + biash[e];
          S[jt][r] = fminf(10.f, fmaxf(-10.f, val));
        }
      }
      // Row softmax. A C-fragment row lives in 16 lanes (xor 1,2,4,8 stays
      // inside each half-wave, which holds distinct rows) x 4 column tiles.
#pragma unroll
      for (int r = 0; r < 8; ++r) {
        float mx = fmaxf(fmaxf(S[0][r], S[1][r]), fmaxf(S[2][r], S[3][r]));
#pragma unroll
        for (int off = 8; off; off >>= 1) mx = fmaxf(mx, __shfl_xor(mx, off, 32));
        float p[4], sum = 0.f;
#pragma unroll
        for (int jt = 0; jt < 4; ++jt) { p[jt] = __expf(S[jt][r] - mx); sum += p[jt]; }
#pragma unroll
        for (int off = 8; off; off >>= 1) sum += __shfl_xor(sum, off, 32);
        const float inv = 1.0f / sum;
        const int i = i_lo + r;
#pragma unroll
        for (int jt = 0; jt < 4; ++jt)
          Ph[i * PS_STRIDE + jt * 16 + jlan] = (bf16_t)(p[jt] * inv);
      }
    }
    // O = P @ V  (intra-wave LDS RAW on Ph: DS ops are in-order per wave)
#pragma unroll 1
    for (int mt = 0; mt < 4; ++mt) {
      const int m0 = mt * 16;
      v16bf pf[2];
      pf[0] = load_frag(Ph, PS_STRIDE, m0, 0, lane);
      pf[1] = load_frag(Ph, PS_STRIDE, m0, 32, lane);
#pragma unroll
      for (int dt = 0; dt < 2; ++dt) {
        v8f acc = {};
#pragma unroll
        for (int ks = 0; ks < 2; ++ks) {
          v16bf vf = load_frag(Vh, VT_STRIDE, dt * 16, ks * 32, lane);
          acc = wmma_bf16(pf[ks], vf, acc);
        }
#pragma unroll
        for (int r = 0; r < 8; ++r) {
          const int m = m0 + r + ((lane >> 4) << 3);
          const int c = h * 32 + dt * 16 + (lane & 15);
          XB[m * XB_STRIDE + c] = (bf16_t)acc[r];   // XB reused as O
        }
      }
    }
  }
  __syncthreads();

  // ---- Stage D: out = O @ proj_w^T + proj_b ------------------------------
  float* og = out + (size_t)b * NTOK * DIMC;
#pragma unroll 1
  for (int mt = 0; mt < 4; ++mt) {
    const int m0 = mt * 16;
    v16bf afr[4];
#pragma unroll
    for (int ks = 0; ks < 4; ++ks) afr[ks] = load_frag(XB, XB_STRIDE, m0, ks * 32, lane);
#pragma unroll 1
    for (int ct = 0; ct < 2; ++ct) {
      const int o0 = wave * 32 + ct * 16;
      v8f acc = {};
#pragma unroll
      for (int ks = 0; ks < 4; ++ks)
        acc = wmma_bf16(afr[ks], load_frag(proj_wb, DIMC, o0, ks * 32, lane), acc);
      const int   o  = o0 + (lane & 15);
      const float pb = proj_b[o];
#pragma unroll
      for (int r = 0; r < 8; ++r) {
        const int m = m0 + r + ((lane >> 4) << 3);
        og[m * DIMC + o] = acc[r] + pb;
      }
    }
  }
}

// ---------------------------------------------------------------------------
extern "C" void kernel_launch(void* const* d_in, const int* in_sizes, int n_in,
                              void* d_out, int out_size, void* d_ws, size_t ws_size,
                              hipStream_t stream) {
  const float* x        = (const float*)d_in[0];
  const float* mask     = (const float*)d_in[1];
  const float* qkv_w    = (const float*)d_in[2];
  const float* qkv_b    = (const float*)d_in[3];
  const float* proj_w   = (const float*)d_in[4];
  const float* proj_b   = (const float*)d_in[5];
  const float* rel_tbl  = (const float*)d_in[6];
  float*       out      = (float*)d_out;

  bf16_t* qkv_wb  = (bf16_t*)((char*)d_ws + WS_QKVW);
  bf16_t* proj_wb = (bf16_t*)((char*)d_ws + WS_PROJW);
  float*  bias_c  = (float*)((char*)d_ws + WS_BIAS);

  wa_prep<<<(3 * 128 * 128 + 255) / 256, 256, 0, stream>>>(
      qkv_w, proj_w, rel_tbl, qkv_wb, proj_wb, bias_c);

  const int num_windows = in_sizes[0] / (NTOK * DIMC);      // 4096
  wa_main<<<num_windows, 128, 0, stream>>>(x, mask, qkv_wb, qkv_b,
                                           proj_wb, proj_b, bias_c, out);
}
